// AttentionBlock_70944269796056
// MI455X (gfx1250) — compile-verified, matrix operand order checked by reading, not run on hardware
//
#include <hip/hip_runtime.h>
#include <math.h>

// ---------------- WMMA fragment types (gfx1250, wave32) ----------------
typedef __attribute__((ext_vector_type(16))) __bf16 v16bf;
typedef __attribute__((ext_vector_type(8)))  float  v8f;
typedef __attribute__((ext_vector_type(4)))  float  f4v;

struct __align__(16) B16 { unsigned int x, y, z, w; };   // 16B POD chunk

union Frag {
  v16bf v;
  B16   q[2];
};

__device__ inline v8f wmma_bf16(const Frag& a, const Frag& b, v8f c) {
  // D = A(16x32 bf16) x B(32x16 bf16) + C(16x16 f32)
  return __builtin_amdgcn_wmma_f32_16x16x32_bf16(
      /*neg_a=*/false, a.v, /*neg_b=*/false, b.v,
      /*c_mod=*/(short)0, c, /*reuse_a=*/false, /*reuse_b=*/false);
}

// ---------------- geometry ----------------
#define TT   32     // tokens per sequence
#define E    256    // embed
#define HD   128    // total head dim
#define NH   8
#define DH   16
#define FF   1024
#define MROW 64     // 2 sequences stacked per workgroup

// ---------------------------------------------------------------------------
// Pack fp32 weight W[K][N] (row-major) into bf16 WMMA-B fragments.
// Tile (nt,kt) covers K rows [kt*32,kt*32+32) x N cols [nt*16,nt*16+16).
// Lane = hh*16+m holds column n = nt*16+m; elements i<8 -> k = hh*8+i,
// i>=8 -> k = 16+hh*8+(i-8). Tile order nt*KT+kt. Mirrors the A-fragment
// K chunking so both fragments load as two 16B vectors per lane.
// ---------------------------------------------------------------------------
__global__ void pack_b_tiles(const float* __restrict__ W, __bf16* __restrict__ Wp,
                             int K, int N) {
  int idx = blockIdx.x * 256 + threadIdx.x;
  int total = K * N;
  if (idx >= total) return;
  int KT   = K >> 5;
  int tile = idx >> 9;          // /512
  int rem  = idx & 511;
  int lane = rem >> 4;
  int i    = rem & 15;
  int nt = tile / KT, kt = tile - nt * KT;
  int hh = lane >> 4, m = lane & 15;
  int kk = (i < 8) ? (hh * 8 + i) : (16 + hh * 8 + (i - 8));
  Wp[idx] = (__bf16)W[(size_t)(kt * 32 + kk) * N + nt * 16 + m];
}

// Quad-M-tile GEMM core: one B fragment feeds four WMMAs
// (rows [0,16),[16,32),[32,48),[48,64) of a 64-row A panel).
template <int KT>
__device__ inline void gemm4(const __bf16* __restrict__ A, int lda,
                             const __bf16* __restrict__ Bp, int nt,
                             int m, int hh, int lane,
                             v8f& c0, v8f& c1, v8f& c2, v8f& c3) {
  for (int kt = 0; kt < KT; ++kt) {
    Frag bfr;
    const __bf16* bp = Bp + (size_t)(nt * KT + kt) * 512 + lane * 16;
    bfr.q[0] = *(const B16*)bp;
    bfr.q[1] = *(const B16*)(bp + 8);
    Frag a0, a1, a2, a3;
    const __bf16* ar = A + (size_t)m * lda + kt * 32;
    a0.q[0] = *(const B16*)(ar + hh * 8);
    a0.q[1] = *(const B16*)(ar + 16 + hh * 8);
    ar += (size_t)16 * lda;
    a1.q[0] = *(const B16*)(ar + hh * 8);
    a1.q[1] = *(const B16*)(ar + 16 + hh * 8);
    ar += (size_t)16 * lda;
    a2.q[0] = *(const B16*)(ar + hh * 8);
    a2.q[1] = *(const B16*)(ar + 16 + hh * 8);
    ar += (size_t)16 * lda;
    a3.q[0] = *(const B16*)(ar + hh * 8);
    a3.q[1] = *(const B16*)(ar + 16 + hh * 8);
    c0 = wmma_bf16(a0, bfr, c0);
    c1 = wmma_bf16(a1, bfr, c1);
    c2 = wmma_bf16(a2, bfr, c2);
    c3 = wmma_bf16(a3, bfr, c3);
  }
}

// ---------------------------------------------------------------------------
// Fused transformer block: one workgroup (8 wave32) per PAIR of sequences.
// All GEMMs run with M=64 (two sequences stacked) so every weight fragment
// loaded from L2 feeds 4 WMMAs.
// ---------------------------------------------------------------------------
__global__ __launch_bounds__(256)
void attn_block_kernel(const float* __restrict__ X,
                       const float* __restrict__ ln_g, const float* __restrict__ ln_b,
                       const __bf16* __restrict__ Wq_p, const float* __restrict__ bq,
                       const __bf16* __restrict__ Wk_p, const float* __restrict__ bk,
                       const __bf16* __restrict__ Wv_p, const float* __restrict__ bv,
                       const __bf16* __restrict__ Wo_p, const float* __restrict__ bo,
                       const __bf16* __restrict__ W1_p, const float* __restrict__ b1,
                       const __bf16* __restrict__ W2_p, const float* __restrict__ b2,
                       float* __restrict__ Out)
{
  __shared__ __align__(16) float  xres[MROW][E];       //  64 KB residual stream
  __shared__ __align__(16) __bf16 hbf [MROW][E];       //  32 KB LN output
  __shared__ __align__(16) __bf16 qbf [MROW][HD];      //  16 KB
  __shared__ __align__(16) __bf16 kbf [MROW][HD];      //  16 KB
  __shared__ __align__(16) __bf16 vtb [2][NH][DH][TT]; //  16 KB V^T per seq/head
  __shared__ __align__(16) __bf16 ctxb[MROW][HD];      //  16 KB attention context
  __shared__ __align__(16) unsigned char bigbuf[131072]; // 128 KB, phase-aliased
  float*  scores = (float*)bigbuf;                  // [16][32][32] f32 (64 KB)
  __bf16* attnbf = (__bf16*)(bigbuf + 65536);       // [16][32][32] bf16 (32 KB)
  __bf16* ffnbf  = (__bf16*)bigbuf;                 // [64][1024] bf16 (128 KB)

  const int tid  = threadIdx.x;
  const int wave = tid >> 5;
  const int lane = tid & 31;
  const int m    = lane & 15;       // fragment column / row-in-half
  const int hh   = lane >> 4;       // lane half selects K/M sub-chunk
  const int b0   = blockIdx.x * 2;  // two adjacent sequences

  const float* Xb = X + (size_t)b0 * TT * E;   // 64 contiguous rows

  // ---- Phase 0: stage X[b0..b0+1] into LDS (coalesced b128, streaming) ----
  {
    const f4v* src = (const f4v*)Xb;
    f4v* dst = (f4v*)&xres[0][0];
    for (int i = tid; i < MROW * E / 4; i += 256)
      dst[i] = __builtin_nontemporal_load(src + i);
  }
  __syncthreads();

  // ---- LayerNorm xres -> hbf (bf16). 8 rows per wave, wave32 shuffles ----
  auto layernorm = [&]() {
    for (int rr = 0; rr < 8; ++rr) {
      int row = wave * 8 + rr;
      float vals[8];
      float s = 0.f, ss = 0.f;
#pragma unroll
      for (int j = 0; j < 8; ++j) {
        float x = xres[row][lane * 8 + j];
        vals[j] = x; s += x; ss += x * x;
      }
#pragma unroll
      for (int off = 16; off > 0; off >>= 1) {
        s  += __shfl_xor(s,  off, 32);
        ss += __shfl_xor(ss, off, 32);
      }
      float mu   = s * (1.f / E);
      float var  = ss * (1.f / E) - mu * mu;
      float rstd = rsqrtf(var + 1e-5f);
#pragma unroll
      for (int j = 0; j < 8; ++j) {
        int c = lane * 8 + j;
        hbf[row][c] = (__bf16)((vals[j] - mu) * rstd * ln_g[c] + ln_b[c]);
      }
    }
  };
  layernorm();
  __syncthreads();

  // ---- Phase 2: QKV projections. wave w owns n-tile w of each of Q, K, V ----
  {
    // Q -> qbf row-major
    float bn = bq[wave * 16 + m];
    v8f c0 = {}, c1 = {}, c2 = {}, c3 = {};
    gemm4<8>(&hbf[0][0], E, Wq_p, wave, m, hh, lane, c0, c1, c2, c3);
#pragma unroll
    for (int j = 0; j < 8; ++j) {
      int r = hh * 8 + j, col = wave * 16 + m;
      qbf[r][col]      = (__bf16)(c0[j] + bn);
      qbf[r + 16][col] = (__bf16)(c1[j] + bn);
      qbf[r + 32][col] = (__bf16)(c2[j] + bn);
      qbf[r + 48][col] = (__bf16)(c3[j] + bn);
    }
  }
  {
    // K -> kbf row-major
    float bn = bk[wave * 16 + m];
    v8f c0 = {}, c1 = {}, c2 = {}, c3 = {};
    gemm4<8>(&hbf[0][0], E, Wk_p, wave, m, hh, lane, c0, c1, c2, c3);
#pragma unroll
    for (int j = 0; j < 8; ++j) {
      int r = hh * 8 + j, col = wave * 16 + m;
      kbf[r][col]      = (__bf16)(c0[j] + bn);
      kbf[r + 16][col] = (__bf16)(c1[j] + bn);
      kbf[r + 32][col] = (__bf16)(c2[j] + bn);
      kbf[r + 48][col] = (__bf16)(c3[j] + bn);
    }
  }
  {
    // V -> vtb transposed [seq][head][dh][token]; 16B packed stores
    float bn = bv[wave * 16 + m];
    v8f c0 = {}, c1 = {}, c2 = {}, c3 = {};
    gemm4<8>(&hbf[0][0], E, Wv_p, wave, m, hh, lane, c0, c1, c2, c3);
    __bf16 t0[8], t1[8], t2[8], t3[8];
#pragma unroll
    for (int j = 0; j < 8; ++j) {
      t0[j] = (__bf16)(c0[j] + bn);   // seq0 tokens hh*8+j
      t1[j] = (__bf16)(c1[j] + bn);   // seq0 tokens 16+hh*8+j
      t2[j] = (__bf16)(c2[j] + bn);   // seq1 tokens hh*8+j
      t3[j] = (__bf16)(c3[j] + bn);   // seq1 tokens 16+hh*8+j
    }
    *(B16*)&vtb[0][wave][m][hh * 8]      = *(const B16*)t0;
    *(B16*)&vtb[0][wave][m][16 + hh * 8] = *(const B16*)t1;
    *(B16*)&vtb[1][wave][m][hh * 8]      = *(const B16*)t2;
    *(B16*)&vtb[1][wave][m][16 + hh * 8] = *(const B16*)t3;
  }
  __syncthreads();

  // ---- Phase 3: attention. 16 (seq,head) tasks, 2 per wave ----
  for (int t = wave; t < 16; t += 8) {
    int seq = t >> 3, head = t & 7;
    const __bf16* Qh = &qbf[seq * TT][head * DH];
    const __bf16* Kh = &kbf[seq * TT][head * DH];
    float*  Sh = scores + t * TT * TT;
    __bf16* Ph = attnbf + t * TT * TT;

    // scores = Qh @ Kh^T (K=16 zero-padded to 32)
    for (int mt = 0; mt < 2; ++mt)
      for (int nt2 = 0; nt2 < 2; ++nt2) {
        Frag a, bf;
        a.q[0]  = *(const B16*)(Qh + (size_t)(mt * 16 + m) * HD + hh * 8);
        a.q[1]  = B16{0, 0, 0, 0};
        bf.q[0] = *(const B16*)(Kh + (size_t)(nt2 * 16 + m) * HD + hh * 8);
        bf.q[1] = B16{0, 0, 0, 0};
        v8f acc = {};
        acc = wmma_bf16(a, bf, acc);
#pragma unroll
        for (int j = 0; j < 8; ++j)
          Sh[(mt * 16 + hh * 8 + j) * TT + nt2 * 16 + m] = acc[j];
      }

    // causal softmax (scale 1/sqrt(16)), lane = query row
    {
      int r = lane;
      float mx = -3.0e38f;
      for (int k = 0; k <= r; ++k) mx = fmaxf(mx, Sh[r * TT + k] * 0.25f);
      float p[TT];
      float sum = 0.f;
#pragma unroll
      for (int k = 0; k < TT; ++k) {
        float pv = (k <= r) ? __expf(Sh[r * TT + k] * 0.25f - mx) : 0.f;
        p[k] = pv; sum += pv;
      }
      float inv = 1.f / sum;
#pragma unroll
      for (int k = 0; k < TT; ++k) Ph[r * TT + k] = (__bf16)(p[k] * inv);
    }

    // ctx = P @ Vh, K=32 single WMMA step (B from transposed V)
    {
      const __bf16* Vh = &vtb[seq][head][0][0];   // [DH][TT]
      Frag bf;
      const __bf16* vr = Vh + (size_t)m * TT;     // row m of Vh^T
      bf.q[0] = *(const B16*)(vr + hh * 8);
      bf.q[1] = *(const B16*)(vr + 16 + hh * 8);
      for (int mt = 0; mt < 2; ++mt) {
        Frag a;
        const __bf16* pr = Ph + (size_t)(mt * 16 + m) * TT;
        a.q[0] = *(const B16*)(pr + hh * 8);
        a.q[1] = *(const B16*)(pr + 16 + hh * 8);
        v8f acc = {};
        acc = wmma_bf16(a, bf, acc);
#pragma unroll
        for (int j = 0; j < 8; ++j)
          ctxb[seq * TT + mt * 16 + hh * 8 + j][head * DH + m] = (__bf16)acc[j];
      }
    }
  }
  __syncthreads();

  // ---- Phase 4: attn_out = ctx @ Wo + bo + X  (residual in xres) ----
  for (int nt = wave * 2; nt < wave * 2 + 2; ++nt) {
    float bn = bo[nt * 16 + m];
    v8f c0 = {}, c1 = {}, c2 = {}, c3 = {};
    gemm4<4>(&ctxb[0][0], HD, Wo_p, nt, m, hh, lane, c0, c1, c2, c3);
#pragma unroll
    for (int j = 0; j < 8; ++j) {
      int r = hh * 8 + j, col = nt * 16 + m;
      xres[r][col]      += c0[j] + bn;
      xres[r + 16][col] += c1[j] + bn;
      xres[r + 32][col] += c2[j] + bn;
      xres[r + 48][col] += c3[j] + bn;
    }
  }
  __syncthreads();

  // ---- Phase 5: second LayerNorm (shared params) ----
  layernorm();
  __syncthreads();

  // ---- Phase 6: FFN1  act = relu(h2 @ W1 + b1)  M=64,K=256,N=1024 ----
  for (int u = 0; u < 8; ++u) {
    int nt = wave * 8 + u;
    float bn = b1[nt * 16 + m];
    v8f c0 = {}, c1 = {}, c2 = {}, c3 = {};
    gemm4<8>(&hbf[0][0], E, W1_p, nt, m, hh, lane, c0, c1, c2, c3);
#pragma unroll
    for (int j = 0; j < 8; ++j) {
      int r = hh * 8 + j, col = nt * 16 + m;
      ffnbf[(size_t)r * FF + col]        = (__bf16)fmaxf(c0[j] + bn, 0.f);
      ffnbf[(size_t)(r + 16) * FF + col] = (__bf16)fmaxf(c1[j] + bn, 0.f);
      ffnbf[(size_t)(r + 32) * FF + col] = (__bf16)fmaxf(c2[j] + bn, 0.f);
      ffnbf[(size_t)(r + 48) * FF + col] = (__bf16)fmaxf(c3[j] + bn, 0.f);
    }
  }
  __syncthreads();

  // ---- Phase 7: out = act @ W2 + b2 + xres  M=64,K=1024,N=256 ----
  float* Ob = Out + (size_t)b0 * TT * E;   // 64 contiguous rows
  for (int nt = wave * 2; nt < wave * 2 + 2; ++nt) {
    float bn = b2[nt * 16 + m];
    v8f c0 = {}, c1 = {}, c2 = {}, c3 = {};
    gemm4<32>(ffnbf, FF, W2_p, nt, m, hh, lane, c0, c1, c2, c3);
#pragma unroll
    for (int j = 0; j < 8; ++j) {
      int r = hh * 8 + j, col = nt * 16 + m;
      __builtin_nontemporal_store(c0[j] + bn + xres[r][col],
                                  Ob + (size_t)r * E + col);
      __builtin_nontemporal_store(c1[j] + bn + xres[r + 16][col],
                                  Ob + (size_t)(r + 16) * E + col);
      __builtin_nontemporal_store(c2[j] + bn + xres[r + 32][col],
                                  Ob + (size_t)(r + 32) * E + col);
      __builtin_nontemporal_store(c3[j] + bn + xres[r + 48][col],
                                  Ob + (size_t)(r + 48) * E + col);
    }
  }
}

// ---------------------------------------------------------------------------
extern "C" void kernel_launch(void* const* d_in, const int* in_sizes, int n_in,
                              void* d_out, int out_size, void* d_ws, size_t ws_size,
                              hipStream_t stream) {
  const float* X    = (const float*)d_in[0];
  const float* ln_g = (const float*)d_in[1];
  const float* ln_b = (const float*)d_in[2];
  const float* Wq   = (const float*)d_in[3];
  const float* bq   = (const float*)d_in[4];
  const float* Wk   = (const float*)d_in[5];
  const float* bk   = (const float*)d_in[6];
  const float* Wv   = (const float*)d_in[7];
  const float* bv   = (const float*)d_in[8];
  const float* Wo   = (const float*)d_in[9];
  const float* bo   = (const float*)d_in[10];
  const float* W1   = (const float*)d_in[11];
  const float* b1   = (const float*)d_in[12];
  const float* W2   = (const float*)d_in[13];
  const float* b2   = (const float*)d_in[14];
  float* Out = (float*)d_out;

  // Packed bf16 weight fragments in workspace (1.25 MB total).
  __bf16* ws   = (__bf16*)d_ws;
  __bf16* Wq_p = ws;              // 256x128  -> 32768 elems
  __bf16* Wk_p = ws + 32768;
  __bf16* Wv_p = ws + 65536;
  __bf16* Wo_p = ws + 98304;      // 128x256  -> 32768
  __bf16* W1_p = ws + 131072;     // 256x1024 -> 262144
  __bf16* W2_p = ws + 393216;     // 1024x256 -> 262144

  pack_b_tiles<<<128,  256, 0, stream>>>(Wq, Wq_p, 256, 128);
  pack_b_tiles<<<128,  256, 0, stream>>>(Wk, Wk_p, 256, 128);
  pack_b_tiles<<<128,  256, 0, stream>>>(Wv, Wv_p, 256, 128);
  pack_b_tiles<<<128,  256, 0, stream>>>(Wo, Wo_p, 128, 256);
  pack_b_tiles<<<1024, 256, 0, stream>>>(W1, W1_p, 256, 1024);
  pack_b_tiles<<<1024, 256, 0, stream>>>(W2, W2_p, 1024, 256);

  attn_block_kernel<<<2048, 256, 0, stream>>>(
      X, ln_g, ln_b,
      Wq_p, bq, Wk_p, bk, Wv_p, bv,
      Wo_p, bo, W1_p, b1, W2_p, b2,
      Out);
}
